// MultiHiddenVITransitionModel_58884001628277
// MI455X (gfx1250) — compile-verified, matrix operand order checked by reading, not run on hardware
//
#include <hip/hip_runtime.h>

namespace {

constexpr int kB   = 256;   // batch
constexpr int kS   = 256;   // state
constexpr int kBel = 1024;  // belief
constexpr int kE   = 1024;  // imu embedding
constexpr int kT   = 32;    // T-1
constexpr int kL   = 11;    // imu sub-sequence

typedef __attribute__((ext_vector_type(16))) __bf16 v16bf;
typedef __attribute__((ext_vector_type(8)))  float  v8f;

__device__ __forceinline__ unsigned short f2bf(float x) {
  union { float f; unsigned int u; } v; v.f = x;
  unsigned int r = v.u + 0x7FFFu + ((v.u >> 16) & 1u);   // round-to-nearest-even
  return (unsigned short)(r >> 16);
}

// One 16-bit WMMA fragment slice for this lane: 8 bf16 at p[0..7] and 8 bf16 at
// p[16..23] (the K and K+16 halves), per ISA 7.12.2 16-bit A/B layouts.
__device__ __forceinline__ v16bf ld_frag(const unsigned short* p) {
  union U { uint4 q[2]; v16bf v; } u;
  u.q[0] = *reinterpret_cast<const uint4*>(p);
  u.q[1] = *reinterpret_cast<const uint4*>(p + 16);
  return u.v;
}

// C[M,N] = A[M,K] @ W[N,K]^T + bias ; optional ReLU ; optional bf16 copy of C.
// M multiple of 64, N multiple of 256, K multiple of 32.
// Block = 8 waves as 2(M) x 4(N); each wave owns a 32x64 strip:
//   2 A fragments x 4 B fragments -> 8 wmma accumulators per k-tile.
// Fragments are explicitly double-buffered so the s_wait_loadcnt before the
// WMMAs covers the *previous* tile's loads (software pipelining).
__global__ __launch_bounds__(256) void gemm_bf16_wmma(
    const unsigned short* __restrict__ A,
    const unsigned short* __restrict__ W,
    const float* __restrict__ bias,
    float* __restrict__ C,
    unsigned short* __restrict__ Cbf,
    int M, int N, int K, int relu) {
  const int lane = threadIdx.x & 31;
  const int wid  = threadIdx.x >> 5;
  const int m0 = blockIdx.x * 64 + (wid >> 2) * 32;    // 2 M-waves
  const int n0 = blockIdx.y * 256 + (wid & 3) * 64;    // 4 N-waves
  const int lr  = lane & 15;
  const int khi = (lane >> 4) * 8;   // lanes 16..31 carry the +8/+24 K halves

  const unsigned short* pa0 = A + (size_t)(m0 + lr) * K + khi;
  const unsigned short* pa1 = pa0 + (size_t)16 * K;
  const unsigned short* pb0 = W + (size_t)(n0 + lr) * K + khi;
  const unsigned short* pb1 = pb0 + (size_t)16 * K;
  const unsigned short* pb2 = pb0 + (size_t)32 * K;
  const unsigned short* pb3 = pb0 + (size_t)48 * K;

  v8f acc[2][4];
#pragma unroll
  for (int i = 0; i < 2; ++i)
#pragma unroll
    for (int j = 0; j < 4; ++j) acc[i][j] = (v8f){};

  // prologue: fragments for k-tile 0
  v16bf a0 = ld_frag(pa0), a1 = ld_frag(pa1);
  v16bf b0 = ld_frag(pb0), b1 = ld_frag(pb1), b2 = ld_frag(pb2), b3 = ld_frag(pb3);

  for (int k = 32; k < K; k += 32) {
    // issue next tile's loads first ...
    v16bf na0 = ld_frag(pa0 + k), na1 = ld_frag(pa1 + k);
    v16bf nb0 = ld_frag(pb0 + k), nb1 = ld_frag(pb1 + k);
    v16bf nb2 = ld_frag(pb2 + k), nb3 = ld_frag(pb3 + k);
    if (k + 32 < K) {                    // pull the tile after that toward L0
      __builtin_prefetch(pb0 + k + 32, 0, 1);
      __builtin_prefetch(pb2 + k + 32, 0, 1);
    }
    // ... then run the current tile's 8 WMMAs while they are in flight
    acc[0][0] = __builtin_amdgcn_wmma_f32_16x16x32_bf16(false, a0, false, b0, (short)0, acc[0][0], false, false);
    acc[0][1] = __builtin_amdgcn_wmma_f32_16x16x32_bf16(false, a0, false, b1, (short)0, acc[0][1], false, false);
    acc[0][2] = __builtin_amdgcn_wmma_f32_16x16x32_bf16(false, a0, false, b2, (short)0, acc[0][2], false, false);
    acc[0][3] = __builtin_amdgcn_wmma_f32_16x16x32_bf16(false, a0, false, b3, (short)0, acc[0][3], false, false);
    acc[1][0] = __builtin_amdgcn_wmma_f32_16x16x32_bf16(false, a1, false, b0, (short)0, acc[1][0], false, false);
    acc[1][1] = __builtin_amdgcn_wmma_f32_16x16x32_bf16(false, a1, false, b1, (short)0, acc[1][1], false, false);
    acc[1][2] = __builtin_amdgcn_wmma_f32_16x16x32_bf16(false, a1, false, b2, (short)0, acc[1][2], false, false);
    acc[1][3] = __builtin_amdgcn_wmma_f32_16x16x32_bf16(false, a1, false, b3, (short)0, acc[1][3], false, false);
    a0 = na0; a1 = na1; b0 = nb0; b1 = nb1; b2 = nb2; b3 = nb3;
  }
  // epilogue tile
  acc[0][0] = __builtin_amdgcn_wmma_f32_16x16x32_bf16(false, a0, false, b0, (short)0, acc[0][0], false, false);
  acc[0][1] = __builtin_amdgcn_wmma_f32_16x16x32_bf16(false, a0, false, b1, (short)0, acc[0][1], false, false);
  acc[0][2] = __builtin_amdgcn_wmma_f32_16x16x32_bf16(false, a0, false, b2, (short)0, acc[0][2], false, false);
  acc[0][3] = __builtin_amdgcn_wmma_f32_16x16x32_bf16(false, a0, false, b3, (short)0, acc[0][3], false, false);
  acc[1][0] = __builtin_amdgcn_wmma_f32_16x16x32_bf16(false, a1, false, b0, (short)0, acc[1][0], false, false);
  acc[1][1] = __builtin_amdgcn_wmma_f32_16x16x32_bf16(false, a1, false, b1, (short)0, acc[1][1], false, false);
  acc[1][2] = __builtin_amdgcn_wmma_f32_16x16x32_bf16(false, a1, false, b2, (short)0, acc[1][2], false, false);
  acc[1][3] = __builtin_amdgcn_wmma_f32_16x16x32_bf16(false, a1, false, b3, (short)0, acc[1][3], false, false);

  // C/D layout: VGPR e -> row (e + 8*(lane>=16)), lane lr -> column.
  const int hi8 = (lane >> 4) * 8;
#pragma unroll
  for (int i = 0; i < 2; ++i) {
    const int rbase = m0 + i * 16 + hi8;
#pragma unroll
    for (int j = 0; j < 4; ++j) {
      const int col = n0 + j * 16 + lr;
      const float bval = bias ? bias[col] : 0.0f;
#pragma unroll
      for (int e = 0; e < 8; ++e) {
        float v = acc[i][j][e] + bval;
        if (relu) v = fmaxf(v, 0.0f);
        const size_t idx = (size_t)(rbase + e) * N + col;
        C[idx] = v;
        if (Cbf) Cbf[idx] = f2bf(v);
      }
    }
  }
}

// fp32 [N,K] weight -> bf16 [N,Kp], zero-padded K.
__global__ void cvt_weight(const float* __restrict__ src,
                           unsigned short* __restrict__ dst,
                           int K, int Kp, int total) {
  int i = blockIdx.x * blockDim.x + threadIdx.x;
  if (i >= total) return;
  int n = i / Kp, k = i - n * Kp;
  dst[i] = f2bf(k < K ? src[(size_t)n * K + k] : 0.0f);
}

// Concatenate up to 3 fp32 row sources into a zero-padded bf16 [B,Kp] matrix.
__global__ void pack_cat(const float* __restrict__ s0, int st0, int w0,
                         const float* __restrict__ s1, int st1, int w1,
                         const float* __restrict__ s2, int st2, int w2,
                         unsigned short* __restrict__ dst, int Kp, int total) {
  int i = blockIdx.x * blockDim.x + threadIdx.x;
  if (i >= total) return;
  int b = i / Kp, k = i - b * Kp;
  float v = 0.0f;
  if (k < w0)                         v = s0[(size_t)b * st0 + k];
  else if (s1 && k < w0 + w1)         v = s1[(size_t)b * st1 + (k - w0)];
  else if (s2 && k < w0 + w1 + w2)    v = s2[(size_t)b * st2 + (k - w0 - w1)];
  dst[i] = f2bf(v);
}

// Fused PyTorch GRUCell gate math; updates h in place, optional bf16 mirror and
// optional fp32 copy into an output slice.
__global__ void gru_fuse(const float* __restrict__ gi, const float* __restrict__ gh,
                         const float* __restrict__ hprev, float* __restrict__ hout,
                         unsigned short* __restrict__ hbf, float* __restrict__ ocopy,
                         int H, int total) {
  int i = blockIdx.x * blockDim.x + threadIdx.x;
  if (i >= total) return;
  int b = i / H, j = i - b * H;
  size_t g = (size_t)b * 3 * H + j;
  float r = 1.0f / (1.0f + expf(-(gi[g] + gh[g])));
  float z = 1.0f / (1.0f + expf(-(gi[g + H] + gh[g + H])));
  float n = tanhf(gi[g + 2 * H] + r * gh[g + 2 * H]);
  float h = (1.0f - z) * n + z * hprev[i];
  hout[i] = h;
  if (hbf)   hbf[i] = f2bf(h);
  if (ocopy) ocopy[i] = h;
}

// v[B,2S] -> mean, std = softplus(raw)+0.1, state = mean + std*noise.
__global__ void head_sample(const float* __restrict__ v, const float* __restrict__ noise,
                            float* __restrict__ o_state, float* __restrict__ o_mean,
                            float* __restrict__ o_std, float* __restrict__ s_next,
                            int total) {
  int i = blockIdx.x * blockDim.x + threadIdx.x;
  if (i >= total) return;
  int b = i / kS, j = i - b * kS;
  float m  = v[(size_t)b * 2 * kS + j];
  float sr = v[(size_t)b * 2 * kS + kS + j];
  float sp = sr > 20.0f ? sr : log1pf(expf(sr));
  float sd = sp + 0.1f;
  float st = m + sd * noise[i];
  o_mean[i]  = m;
  o_std[i]   = sd;
  o_state[i] = st;
  s_next[i]  = st;
}

}  // namespace

extern "C" void kernel_launch(void* const* d_in, const int* in_sizes, int n_in,
                              void* d_out, int out_size, void* d_ws, size_t ws_size,
                              hipStream_t stream) {
  (void)in_sizes; (void)n_in; (void)out_size;

  const float* prev_state  = (const float*)d_in[0];
  const float* poses       = (const float*)d_in[1];
  const float* prev_belief = (const float*)d_in[2];
  const float* obs_v       = (const float*)d_in[3];
  const float* obs_i       = (const float*)d_in[4];
  const float* noise_post  = (const float*)d_in[5];
  const float* noise_prior = (const float*)d_in[6];
  const float* Wv = (const float*)d_in[7],  * bv = (const float*)d_in[8];
  const float* Wi = (const float*)d_in[9],  * bi = (const float*)d_in[10];
  const float* Wp = (const float*)d_in[11], * bp = (const float*)d_in[12];
  const float* gvWih = (const float*)d_in[13], * gvWhh = (const float*)d_in[14];
  const float* gvbih = (const float*)d_in[15], * gvbhh = (const float*)d_in[16];
  const float* giWih = (const float*)d_in[17], * giWhh = (const float*)d_in[18];
  const float* gibih = (const float*)d_in[19], * gibhh = (const float*)d_in[20];
  const float* gpWih = (const float*)d_in[21], * gpWhh = (const float*)d_in[22];
  const float* gpbih = (const float*)d_in[23], * gpbhh = (const float*)d_in[24];
  const float* i0Wih = (const float*)d_in[25], * i0Whh = (const float*)d_in[26];
  const float* i0bih = (const float*)d_in[27], * i0bhh = (const float*)d_in[28];
  const float* i1Wih = (const float*)d_in[29], * i1Whh = (const float*)d_in[30];
  const float* i1bih = (const float*)d_in[31], * i1bhh = (const float*)d_in[32];
  const float* Wbp  = (const float*)d_in[33], * bbp  = (const float*)d_in[34];
  const float* Wsp  = (const float*)d_in[35], * bsp  = (const float*)d_in[36];
  const float* Wbpr = (const float*)d_in[37], * bbpr = (const float*)d_in[38];
  const float* Wspr = (const float*)d_in[39], * bspr = (const float*)d_in[40];

  // -------- workspace carve --------
  size_t off = 0;
  auto carve = [&](size_t bytes) -> void* {
    off = (off + 255) & ~(size_t)255;
    void* p = (char*)d_ws + off;
    off += bytes;
    return p;
  };
  auto bfb = [&](size_t n) { return (unsigned short*)carve(n * 2); };
  auto f32 = [&](size_t n) { return (float*)carve(n * 4); };

  // bf16 weight images (K padded to multiple of 32)
  unsigned short* wWv    = bfb((size_t)1024 * 1536);
  unsigned short* wWi    = bfb((size_t)1024 * 1536);
  unsigned short* wWp    = bfb((size_t)1024 * 544);   // 518 -> 544
  unsigned short* wgvWih = bfb((size_t)3072 * 1024);
  unsigned short* wgvWhh = bfb((size_t)3072 * 1024);
  unsigned short* wgiWih = bfb((size_t)3072 * 1024);
  unsigned short* wgiWhh = bfb((size_t)3072 * 1024);
  unsigned short* wgpWih = bfb((size_t)3072 * 1024);
  unsigned short* wgpWhh = bfb((size_t)3072 * 1024);
  unsigned short* wi0Wih = bfb((size_t)3072 * 32);    // 6 -> 32
  unsigned short* wi0Whh = bfb((size_t)3072 * 1024);
  unsigned short* wi1Wih = bfb((size_t)3072 * 1024);
  unsigned short* wi1Whh = bfb((size_t)3072 * 1024);
  unsigned short* wWbp   = bfb((size_t)1024 * 2048);
  unsigned short* wWsp   = bfb((size_t)512 * 1024);
  unsigned short* wWbpr  = bfb((size_t)1024 * 1024);
  unsigned short* wWspr  = bfb((size_t)512 * 1024);

  // fp32 recurrent state + scratch
  float* post_c  = f32((size_t)kB * kS);
  float* prior_c = f32((size_t)kB * kS);
  float* post_n  = f32((size_t)kB * kS);
  float* prior_n = f32((size_t)kB * kS);
  float* pbv = f32((size_t)kB * kBel);
  float* pbi = f32((size_t)kB * kBel);
  float* pb  = f32((size_t)kB * kBel);
  float* ih0 = f32((size_t)kB * kE);
  float* ih1 = f32((size_t)kB * kE);
  float* h_f32  = f32((size_t)kB * 1024);
  float* gi_buf = f32((size_t)kB * 3072);
  float* gh_buf = f32((size_t)kB * 3072);
  float* headv  = f32((size_t)kB * 512);

  // bf16 activation mirrors / staging
  unsigned short* catbf  = bfb((size_t)kB * 2048);
  unsigned short* h_bf   = bfb((size_t)kB * 1024);
  unsigned short* pbv_bf = bfb((size_t)kB * kBel);
  unsigned short* pbi_bf = bfb((size_t)kB * kBel);
  unsigned short* pb_bf  = bfb((size_t)kB * kBel);
  unsigned short* h0_bf  = bfb((size_t)kB * kE);
  unsigned short* h1_bf  = bfb((size_t)kB * kE);

  if (off > ws_size) return;  // workspace too small: do nothing (deterministic)

  // -------- launch helpers --------
  auto cvt = [&](const float* src, unsigned short* dst, int N, int K, int Kp) {
    int total = N * Kp;
    cvt_weight<<<(total + 255) / 256, 256, 0, stream>>>(src, dst, K, Kp, total);
  };
  auto gemm = [&](const unsigned short* A, const unsigned short* W, const float* bias,
                  float* C, unsigned short* Cbf, int N, int K, int relu) {
    dim3 grid(kB / 64, N / 256);
    gemm_bf16_wmma<<<grid, 256, 0, stream>>>(A, W, bias, C, Cbf, kB, N, K, relu);
  };
  auto pack = [&](const float* s0, int st0, int w0, const float* s1, int st1, int w1,
                  const float* s2, int st2, int w2, unsigned short* dst, int Kp) {
    int total = kB * Kp;
    pack_cat<<<(total + 255) / 256, 256, 0, stream>>>(s0, st0, w0, s1, st1, w1,
                                                      s2, st2, w2, dst, Kp, total);
  };
  auto gru = [&](float* h, unsigned short* hbf, float* ocopy) {
    int total = kB * 1024;
    gru_fuse<<<(total + 255) / 256, 256, 0, stream>>>(gi_buf, gh_buf, h, h, hbf,
                                                      ocopy, 1024, total);
  };
  auto head = [&](const float* noise_t, float* ost, float* omn, float* osd, float* snext) {
    int total = kB * kS;
    head_sample<<<(total + 255) / 256, 256, 0, stream>>>(headv, noise_t, ost, omn,
                                                         osd, snext, total);
  };

  // -------- weight conversion (every call; bf16 set ~73MB, L2-resident) --------
  cvt(Wv, wWv, 1024, 1536, 1536);   cvt(Wi, wWi, 1024, 1536, 1536);
  cvt(Wp, wWp, 1024, 518, 544);
  cvt(gvWih, wgvWih, 3072, 1024, 1024); cvt(gvWhh, wgvWhh, 3072, 1024, 1024);
  cvt(giWih, wgiWih, 3072, 1024, 1024); cvt(giWhh, wgiWhh, 3072, 1024, 1024);
  cvt(gpWih, wgpWih, 3072, 1024, 1024); cvt(gpWhh, wgpWhh, 3072, 1024, 1024);
  cvt(i0Wih, wi0Wih, 3072, 6, 32);      cvt(i0Whh, wi0Whh, 3072, 1024, 1024);
  cvt(i1Wih, wi1Wih, 3072, 1024, 1024); cvt(i1Whh, wi1Whh, 3072, 1024, 1024);
  cvt(Wbp, wWbp, 1024, 2048, 2048);     cvt(Wsp, wWsp, 512, 1024, 1024);
  cvt(Wbpr, wWbpr, 1024, 1024, 1024);   cvt(Wspr, wWspr, 512, 1024, 1024);

  // -------- state init --------
  hipMemcpyAsync(post_c,  prev_state,  (size_t)kB * kS * 4, hipMemcpyDeviceToDevice, stream);
  hipMemcpyAsync(prior_c, prev_state,  (size_t)kB * kS * 4, hipMemcpyDeviceToDevice, stream);
  hipMemcpyAsync(pbv, prev_belief, (size_t)kB * kBel * 4, hipMemcpyDeviceToDevice, stream);
  hipMemcpyAsync(pbi, prev_belief, (size_t)kB * kBel * 4, hipMemcpyDeviceToDevice, stream);
  hipMemcpyAsync(pb,  prev_belief, (size_t)kB * kBel * 4, hipMemcpyDeviceToDevice, stream);
  hipMemsetAsync(ih0, 0, (size_t)kB * kE * 4, stream);
  hipMemsetAsync(ih1, 0, (size_t)kB * kE * 4, stream);
  hipMemsetAsync(h0_bf, 0, (size_t)kB * kE * 2, stream);  // bf16 zero == 0x0000
  hipMemsetAsync(h1_bf, 0, (size_t)kB * kE * 2, stream);
  pack(prev_belief, kBel, kBel, nullptr, 0, 0, nullptr, 0, 0, pbv_bf, kBel);
  pack(prev_belief, kBel, kBel, nullptr, 0, 0, nullptr, 0, 0, pbi_bf, kBel);
  pack(prev_belief, kBel, kBel, nullptr, 0, 0, nullptr, 0, 0, pb_bf,  kBel);

  // -------- output slices --------
  float* out = (float*)d_out;
  const size_t NB = (size_t)kT * kB * kBel;
  const size_t NS = (size_t)kT * kB * kS;
  float* o_pbv  = out;
  float* o_pbi  = o_pbv + NB;
  float* o_prs  = o_pbi + NB;
  float* o_prm  = o_prs + NS;
  float* o_prsd = o_prm + NS;
  float* o_pos  = o_prsd + NS;
  float* o_pom  = o_pos + NS;
  float* o_posd = o_pom + NS;

  // -------- time scan --------
  for (int t = 0; t < kT; ++t) {
    const float* ov_t   = obs_v + (size_t)t * kB * kE;
    const float* pose_t = poses + (size_t)t * kB * 6;
    const size_t tBS = (size_t)t * kB * kS;
    const size_t tBB = (size_t)t * kB * kBel;

    // ---- visual posterior belief ----
    pack(ov_t, kE, kE, post_c, kS, kS, prior_c, kS, kS, catbf, 1536);
    gemm(catbf, wWv, bv, h_f32, h_bf, 1024, 1536, 1);
    gemm(h_bf,   wgvWih, gvbih, gi_buf, nullptr, 3072, 1024, 0);
    gemm(pbv_bf, wgvWhh, gvbhh, gh_buf, nullptr, 3072, 1024, 0);
    gru(pbv, pbv_bf, o_pbv + tBB);

    // ---- 2-layer IMU GRU over L=11 sub-steps ----
    for (int l = 0; l < kL; ++l) {
      const float* x_l = obs_i + (size_t)t * kB * kL * 6 + (size_t)l * 6;  // [B,6], stride L*6
      pack(x_l, kL * 6, 6, nullptr, 0, 0, nullptr, 0, 0, catbf, 32);
      gemm(catbf, wi0Wih, i0bih, gi_buf, nullptr, 3072, 32, 0);
      gemm(h0_bf, wi0Whh, i0bhh, gh_buf, nullptr, 3072, 1024, 0);
      gru(ih0, h0_bf, nullptr);                       // h0 <- layer0 cell
      gemm(h0_bf, wi1Wih, i1bih, gi_buf, nullptr, 3072, 1024, 0);
      gemm(h1_bf, wi1Whh, i1bhh, gh_buf, nullptr, 3072, 1024, 0);
      gru(ih1, h1_bf, nullptr);                       // h1 <- layer1 cell
    }

    // ---- imu posterior belief (uses ys1[-1] == ih1) ----
    pack(ih1, kE, kE, post_c, kS, kS, prior_c, kS, kS, catbf, 1536);
    gemm(catbf, wWi, bi, h_f32, h_bf, 1024, 1536, 1);
    gemm(h_bf,   wgiWih, gibih, gi_buf, nullptr, 3072, 1024, 0);
    gemm(pbi_bf, wgiWhh, gibhh, gh_buf, nullptr, 3072, 1024, 0);
    gru(pbi, pbi_bf, o_pbi + tBB);

    // ---- posterior head (uses NEW pbv, pbi) ----
    pack(pbv, kBel, kBel, pbi, kBel, kBel, nullptr, 0, 0, catbf, 2048);
    gemm(catbf, wWbp, bbp, h_f32, h_bf, 1024, 2048, 1);
    gemm(h_bf, wWsp, bsp, headv, nullptr, 512, 1024, 0);
    head(noise_post + tBS, o_pos + tBS, o_pom + tBS, o_posd + tBS, post_n);

    // ---- prior branch (uses OLD post_c / prior_c) ----
    pack(pose_t, 6, 6, post_c, kS, kS, prior_c, kS, kS, catbf, 544);
    gemm(catbf, wWp, bp, h_f32, h_bf, 1024, 544, 1);
    gemm(h_bf,  wgpWih, gpbih, gi_buf, nullptr, 3072, 1024, 0);
    gemm(pb_bf, wgpWhh, gpbhh, gh_buf, nullptr, 3072, 1024, 0);
    gru(pb, pb_bf, nullptr);
    gemm(pb_bf, wWbpr, bbpr, h_f32, h_bf, 1024, 1024, 1);
    gemm(h_bf, wWspr, bspr, headv, nullptr, 512, 1024, 0);
    head(noise_prior + tBS, o_prs + tBS, o_prm + tBS, o_prsd + tBS, prior_n);

    // ---- commit sampled states ----
    float* tmp;
    tmp = post_c;  post_c  = post_n;  post_n  = tmp;
    tmp = prior_c; prior_c = prior_n; prior_n = tmp;
  }
}
